// GNN_SAGEConv_LSTM_39410619908364
// MI455X (gfx1250) — compile-verified, matrix operand order checked by reading, not run on hardware
//
#include <hip/hip_runtime.h>
#include <hip/hip_bf16.h>
#include <stdint.h>

// ---------------------------------------------------------------------------
// Types for CDNA5 WMMA (wave32, v_wmma_f32_16x16x32_f16)
// ---------------------------------------------------------------------------
typedef _Float16 h16_t;
typedef __attribute__((ext_vector_type(16))) _Float16 v16h;
typedef __attribute__((ext_vector_type(8)))  _Float16 v8h;
typedef __attribute__((ext_vector_type(8)))  float    v8f;

static constexpr int N_NODES = 20000;
static constexpr int DEG     = 16;
static constexpr int HID     = 256;   // out_c of both GC layers
static constexpr int G4      = 1024;  // 4*HID
static constexpr int NGRAPH  = 64;

// ---------------------------------------------------------------------------
// Fragment loads per ISA 16-bit A-matrix 16x32 layout:
//   lane l (li=l&15, hf=l>>4) holds row/col li, K chunks:
//   chunk0 = [kb + 8*hf, +8), chunk1 = [kb + 16 + 8*hf, +8)
// B (32x16) uses the symmetric layout with li = column, reading row li of W
// (W stored row-major [Ncols x K], i.e. column li of W^T is row li of W).
// ---------------------------------------------------------------------------
__device__ __forceinline__ v16h ldfrag(const h16_t* __restrict__ row, int kb, int hf) {
    v8h c0 = *(const v8h*)(row + kb + (hf << 3));
    v8h c1 = *(const v8h*)(row + kb + 16 + (hf << 3));
    v16h r;
#pragma unroll
    for (int i = 0; i < 8; ++i) { r[i] = c0[i]; r[i + 8] = c1[i]; }
    return r;
}

__device__ __forceinline__ v8f wmma16(v16h a, v16h b, v8f c) {
    return __builtin_amdgcn_wmma_f32_16x16x32_f16(
        /*neg_a=*/false, a, /*neg_b=*/false, b,
        /*c_mod=*/(short)0, c, /*reuse_a=*/false, /*reuse_b=*/false);
}

__device__ __forceinline__ float sigmoidf(float x) { return 1.0f / (1.0f + __expf(-x)); }

// ---------------------------------------------------------------------------
// Generic WMMA GEMM: C[N x NC] (+)= A[N x K] @ W^T  (+ bias), W is [NC x K].
// Block = 256 threads = 8 waves; block tile = 128 rows x 64 cols;
// wave tile = 16 rows x 64 cols (4 accumulators).
// ---------------------------------------------------------------------------
template <bool OUT_F16, bool ACC>
__global__ void gemm_wmma(const h16_t* __restrict__ A, const h16_t* __restrict__ W,
                          const float* __restrict__ bias, void* __restrict__ Cv,
                          int N, int K, int NC) {
    const int wave = threadIdx.x >> 5;
    const int lane = threadIdx.x & 31;
    const int li = lane & 15, hf = lane >> 4;
    const int row0 = blockIdx.y * 128 + wave * 16;
    if (row0 >= N) return;
    const int col0 = blockIdx.x * 64;

    v8f acc[4];
#pragma unroll
    for (int c = 0; c < 4; ++c) acc[c] = (v8f)(0.0f);

    const h16_t* arow = A + (size_t)(row0 + li) * K;
    const h16_t* wrow = W + (size_t)(col0 + li) * K;
    for (int kb = 0; kb < K; kb += 32) {
        v16h af = ldfrag(arow, kb, hf);
#pragma unroll
        for (int c = 0; c < 4; ++c) {
            v16h bf = ldfrag(wrow + (size_t)(c * 16) * K, kb, hf);
            acc[c] = wmma16(af, bf, acc[c]);
        }
    }
#pragma unroll
    for (int c = 0; c < 4; ++c) {
        const int n = col0 + c * 16 + li;
        const float bv = bias ? bias[n] : 0.0f;
#pragma unroll
        for (int r = 0; r < 8; ++r) {
            const int m = row0 + r + (hf << 3);
            const size_t idx = (size_t)m * NC + n;
            float v = acc[c][r] + bv;
            if constexpr (OUT_F16) {
                h16_t* C = (h16_t*)Cv;
                if constexpr (ACC) v += (float)C[idx];
                C[idx] = (h16_t)v;
            } else {
                float* C = (float*)Cv;
                if constexpr (ACC) v += C[idx];
                C[idx] = v;
            }
        }
    }
}

// ---------------------------------------------------------------------------
// Fused LSTM step:  gates = hin @ Whh^T + gather(Y, src[:,t]) + (b_ih+b_hh)
// then c = sig(f)*c + sig(i)*tanh(g);  hout = sig(o)*tanh(c).
// Wave tile: 16 nodes x 16 hidden units; computes all 4 gate columns
// (i/f/g/o) for its units so the elementwise update is fully in-register.
// h double-buffered across steps (hin/hout); c updated in place (disjoint).
// ---------------------------------------------------------------------------
__global__ void lstm_step(const h16_t* __restrict__ Whh,   // [1024 x 256]
                          const h16_t* __restrict__ Y,     // [N x 1024]
                          const float* __restrict__ bsum,  // [1024] = b_ih+b_hh
                          const int*   __restrict__ src,   // [N*DEG]
                          const h16_t* __restrict__ hin,   // [N x 256]
                          h16_t*       __restrict__ hout,  // [N x 256]
                          float*       __restrict__ cst,   // [N x 256]
                          int t, int N) {
    const int wave = threadIdx.x >> 5;
    const int lane = threadIdx.x & 31;
    const int li = lane & 15, hf = lane >> 4;
    const int u0   = blockIdx.x * 128 + wave * 16;  // hidden-unit tile
    const int row0 = blockIdx.y * 16;               // node tile

    v8f acc[4];
#pragma unroll
    for (int g = 0; g < 4; ++g) acc[g] = (v8f)(0.0f);

    const h16_t* arow = hin + (size_t)(row0 + li) * HID;
#pragma unroll
    for (int kb = 0; kb < HID; kb += 32) {
        v16h af = ldfrag(arow, kb, hf);
#pragma unroll
        for (int g = 0; g < 4; ++g) {
            const h16_t* wr = Whh + (size_t)(g * HID + u0 + li) * HID;
            acc[g] = wmma16(af, ldfrag(wr, kb, hf), acc[g]);
        }
    }

    const int u = u0 + li;
    const float bi = bsum[u], bf_ = bsum[HID + u], bg = bsum[2 * HID + u], bo = bsum[3 * HID + u];
#pragma unroll
    for (int r = 0; r < 8; ++r) {
        const int node = row0 + r + (hf << 3);
        const int s = src[node * DEG + t];
        const h16_t* yrow = Y + (size_t)s * G4;
        const float gi = sigmoidf(acc[0][r] + (float)yrow[u]           + bi);
        const float gf = sigmoidf(acc[1][r] + (float)yrow[HID + u]     + bf_);
        const float gg = tanhf   (acc[2][r] + (float)yrow[2 * HID + u] + bg);
        const float go = sigmoidf(acc[3][r] + (float)yrow[3 * HID + u] + bo);
        const size_t idx = (size_t)node * HID + u;
        const float c = gf * cst[idx] + gi * gg;
        cst[idx] = c;
        hout[idx] = (h16_t)(go * tanhf(c));
    }
}

// ---------------------------------------------------------------------------
// Small helper kernels
// ---------------------------------------------------------------------------
__global__ void cvt_f32_to_f16(const float* __restrict__ in, h16_t* __restrict__ out, size_t n) {
    for (size_t i = (size_t)blockIdx.x * blockDim.x + threadIdx.x; i < n;
         i += (size_t)gridDim.x * blockDim.x)
        out[i] = (h16_t)in[i];
}

__global__ void fill_zero_u32(uint32_t* __restrict__ p, size_t n) {
    for (size_t i = (size_t)blockIdx.x * blockDim.x + threadIdx.x; i < n;
         i += (size_t)gridDim.x * blockDim.x)
        p[i] = 0u;
}

__global__ void add_vec(const float* __restrict__ a, const float* __restrict__ b,
                        float* __restrict__ o, int n) {
    int i = blockIdx.x * blockDim.x + threadIdx.x;
    if (i < n) o[i] = a[i] + b[i];
}

// Column sums/sumsq over [N x 256] f32 (coalesced: thread = column)
__global__ void bn_partial(const float* __restrict__ C, float* __restrict__ sum,
                           float* __restrict__ sq, int N, int rowsPerBlock) {
    const int col = threadIdx.x;
    const int r0 = blockIdx.x * rowsPerBlock;
    int r1 = r0 + rowsPerBlock; if (r1 > N) r1 = N;
    float s = 0.0f, q = 0.0f;
    for (int r = r0; r < r1; ++r) {
        float v = C[(size_t)r * HID + col];
        s += v; q += v * v;
    }
    atomicAdd(&sum[col], s);
    atomicAdd(&sq[col], q);
}

// normalize + leaky; emit f16 (next layer input) and f32 (pooling source)
__global__ void bn_apply(const float* __restrict__ C, const float* __restrict__ sum,
                         const float* __restrict__ sq, const float* __restrict__ g,
                         const float* __restrict__ b, h16_t* __restrict__ o16,
                         float* __restrict__ o32, size_t total, float invN) {
    for (size_t i = (size_t)blockIdx.x * blockDim.x + threadIdx.x; i < total;
         i += (size_t)gridDim.x * blockDim.x) {
        const int col = (int)(i & (HID - 1));
        const float m = sum[col] * invN;
        const float var = sq[col] * invN - m * m;
        float v = (C[i] - m) * rsqrtf(var + 1e-5f) * g[col] + b[col];
        v = (v >= 0.0f) ? v : 0.01f * v;
        o16[i] = (h16_t)v;
        o32[i] = v;
    }
}

__global__ void pool_count(const int* __restrict__ batch, float* __restrict__ counts, int N) {
    int i = blockIdx.x * blockDim.x + threadIdx.x;
    if (i < N) atomicAdd(&counts[batch[i]], 1.0f);
}

__global__ void pool_add(const float* __restrict__ h, const int* __restrict__ batch,
                         float* __restrict__ pooled, int N) {
    size_t i = (size_t)blockIdx.x * blockDim.x + threadIdx.x;
    if (i < (size_t)N * HID) {
        const int node = (int)(i >> 8);
        const int c = (int)(i & (HID - 1));
        atomicAdd(&pooled[(size_t)batch[node] * HID + c], h[i]);
    }
}

__global__ void pool_div(float* __restrict__ pooled, const float* __restrict__ counts) {
    const int g = blockIdx.x, c = threadIdx.x;
    pooled[(size_t)g * HID + c] /= fmaxf(counts[g], 1.0f);
}

// tiny dense layer: out[R x U] = X[R x K] @ W^T + b   (R=64)
__global__ void dense_small(const float* __restrict__ X, const float* __restrict__ W,
                            const float* __restrict__ b, float* __restrict__ out,
                            int K, int U) {
    const int row = blockIdx.x, u = threadIdx.x;
    if (u >= U) return;
    const float* xr = X + (size_t)row * K;
    const float* wr = W + (size_t)u * K;
    float s = b[u];
    for (int k = 0; k < K; ++k) s += xr[k] * wr[k];
    out[(size_t)row * U + u] = s;
}

// in-place BN (over 64 rows) + leaky for the dense head, one block
__global__ void bn_leaky_small(float* __restrict__ X, const float* __restrict__ g,
                               const float* __restrict__ b, int R, int U) {
    const int u = threadIdx.x;
    if (u >= U) return;
    float s = 0.0f, q = 0.0f;
    for (int r = 0; r < R; ++r) { float v = X[(size_t)r * U + u]; s += v; q += v * v; }
    const float m = s / R;
    const float var = q / R - m * m;
    const float rs = rsqrtf(var + 1e-5f);
    for (int r = 0; r < R; ++r) {
        float v = (X[(size_t)r * U + u] - m) * rs * g[u] + b[u];
        X[(size_t)r * U + u] = (v >= 0.0f) ? v : 0.01f * v;
    }
}

__global__ void reg_head(const float* __restrict__ X, const float* __restrict__ W,
                         const float* __restrict__ b, float* __restrict__ out, int K) {
    const int i = threadIdx.x;  // 64 graphs
    float s = b[0];
    const float* xr = X + (size_t)i * K;
    for (int k = 0; k < K; ++k) s += xr[k] * W[k];
    out[i] = s;
}

// ---------------------------------------------------------------------------
// Host-side orchestration
// ---------------------------------------------------------------------------
extern "C" void kernel_launch(void* const* d_in, const int* in_sizes, int n_in,
                              void* d_out, int out_size, void* d_ws, size_t ws_size,
                              hipStream_t stream) {
    (void)in_sizes; (void)n_in; (void)out_size; (void)ws_size;
    const int N = N_NODES;

    const float* x     = (const float*)d_in[0];
    const int*   eidx  = (const int*)d_in[1];   // [2, E]; row 0 = src (dst already sorted)
    const int*   batch = (const int*)d_in[2];
    const int*   src   = eidx;                  // argsort(dst) is identity

    // ---- workspace bump allocator ----
    char* ws = (char*)d_ws;
    size_t off = 0;
    auto bump = [&](size_t bytes) -> char* {
        char* p = ws + off;
        off += (bytes + 255) & ~(size_t)255;
        return p;
    };
    h16_t* h16buf[2]; h16buf[0] = (h16_t*)bump((size_t)N * HID * 2);
    h16buf[1] = (h16_t*)bump((size_t)N * HID * 2);
    h16_t* y16   = (h16_t*)bump((size_t)N * G4 * 2);
    h16_t* hs16[2]; hs16[0] = (h16_t*)bump((size_t)N * HID * 2);
    hs16[1] = (h16_t*)bump((size_t)N * HID * 2);
    float* cst   = (float*)bump((size_t)N * HID * 4);
    float* cf32  = (float*)bump((size_t)N * HID * 4);
    float* hf32  = (float*)bump((size_t)N * HID * 4);
    h16_t* wih16 = (h16_t*)bump((size_t)G4 * HID * 2);
    h16_t* whh16 = (h16_t*)bump((size_t)G4 * HID * 2);
    h16_t* wl16  = (h16_t*)bump((size_t)HID * HID * 2);
    h16_t* wr16  = (h16_t*)bump((size_t)HID * HID * 2);
    float* bsum  = (float*)bump((size_t)G4 * 4);
    float* bnst  = (float*)bump((size_t)2 * HID * 4);   // sums | sumsq
    float* pooled = (float*)bump((size_t)NGRAPH * HID * 4);
    float* counts = (float*)bump((size_t)NGRAPH * 4);
    float* dbuf0  = (float*)bump((size_t)NGRAPH * HID * 4);
    float* dbuf1  = (float*)bump((size_t)NGRAPH * HID * 4);

    const dim3 B256(256);

    // x -> f16
    cvt_f32_to_f16<<<dim3(2048), B256, 0, stream>>>(x, h16buf[0], (size_t)N * 128);

    int cur = 0;
    int in_c = 128;
    for (int l = 0; l < 2; ++l) {
        const int base = 3 + l * 9;
        const float* W_ih = (const float*)d_in[base + 0];   // [1024 x in_c]
        const float* W_hh = (const float*)d_in[base + 1];   // [1024 x 256]
        const float* b_ih = (const float*)d_in[base + 2];
        const float* b_hh = (const float*)d_in[base + 3];
        const float* W_l  = (const float*)d_in[base + 4];   // [256 x 256]
        const float* b_l  = (const float*)d_in[base + 5];
        const float* W_r  = (const float*)d_in[base + 6];   // [256 x in_c]
        const float* bn_g = (const float*)d_in[base + 7];
        const float* bn_b = (const float*)d_in[base + 8];

        cvt_f32_to_f16<<<dim3(1024), B256, 0, stream>>>(W_ih, wih16, (size_t)G4 * in_c);
        cvt_f32_to_f16<<<dim3(1024), B256, 0, stream>>>(W_hh, whh16, (size_t)G4 * HID);
        cvt_f32_to_f16<<<dim3(256),  B256, 0, stream>>>(W_l, wl16, (size_t)HID * HID);
        cvt_f32_to_f16<<<dim3(256),  B256, 0, stream>>>(W_r, wr16, (size_t)HID * in_c);
        add_vec<<<dim3(4), B256, 0, stream>>>(b_ih, b_hh, bsum, G4);

        // Y = h @ W_ih^T   (f16 out, L2-resident)
        {
            dim3 g(G4 / 64, (N + 127) / 128);
            gemm_wmma<true, false><<<g, B256, 0, stream>>>(h16buf[cur], wih16, nullptr,
                                                           (void*)y16, N, in_c, G4);
        }

        // zero h0, c0
        fill_zero_u32<<<dim3(4096), B256, 0, stream>>>((uint32_t*)hs16[0], (size_t)N * HID / 2);
        fill_zero_u32<<<dim3(4096), B256, 0, stream>>>((uint32_t*)cst, (size_t)N * HID);

        // 16 recurrent steps (WMMA-fused gate GEMM + activation update)
        for (int t = 0; t < DEG; ++t) {
            dim3 g(2, N / 16);
            lstm_step<<<g, B256, 0, stream>>>(whh16, y16, bsum, src,
                                              hs16[t & 1], hs16[(t + 1) & 1], cst, t, N);
        }
        h16_t* agg = hs16[0];  // after t=15, output lives in buffer 0

        // C = agg @ W_l^T + b_l ; C += h_prev @ W_r^T
        {
            dim3 g(HID / 64, (N + 127) / 128);
            gemm_wmma<false, false><<<g, B256, 0, stream>>>(agg, wl16, b_l,
                                                            (void*)cf32, N, HID, HID);
            gemm_wmma<false, true><<<g, B256, 0, stream>>>(h16buf[cur], wr16, nullptr,
                                                           (void*)cf32, N, in_c, HID);
        }

        // BatchNorm + LeakyReLU
        fill_zero_u32<<<dim3(2), B256, 0, stream>>>((uint32_t*)bnst, 2 * HID);
        bn_partial<<<dim3(100), B256, 0, stream>>>(cf32, bnst, bnst + HID, N, 200);
        bn_apply<<<dim3(8192), B256, 0, stream>>>(cf32, bnst, bnst + HID, bn_g, bn_b,
                                                  h16buf[1 - cur], hf32,
                                                  (size_t)N * HID, 1.0f / (float)N);
        cur = 1 - cur;
        in_c = HID;
    }

    // ---- graph mean pooling ----
    fill_zero_u32<<<dim3(64), B256, 0, stream>>>((uint32_t*)pooled, (size_t)NGRAPH * HID);
    fill_zero_u32<<<dim3(1), B256, 0, stream>>>((uint32_t*)counts, NGRAPH);
    pool_count<<<dim3((N + 255) / 256), B256, 0, stream>>>(batch, counts, N);
    pool_add<<<dim3((N * HID + 255) / 256), B256, 0, stream>>>(hf32, batch, pooled, N);
    pool_div<<<dim3(NGRAPH), dim3(HID), 0, stream>>>(pooled, counts);

    // ---- dense head ----
    dense_small<<<dim3(NGRAPH), dim3(256), 0, stream>>>(pooled, (const float*)d_in[21],
                                                        (const float*)d_in[22], dbuf0, 256, 256);
    bn_leaky_small<<<dim3(1), dim3(256), 0, stream>>>(dbuf0, (const float*)d_in[23],
                                                      (const float*)d_in[24], NGRAPH, 256);
    dense_small<<<dim3(NGRAPH), dim3(128), 0, stream>>>(dbuf0, (const float*)d_in[25],
                                                        (const float*)d_in[26], dbuf1, 256, 128);
    bn_leaky_small<<<dim3(1), dim3(128), 0, stream>>>(dbuf1, (const float*)d_in[27],
                                                      (const float*)d_in[28], NGRAPH, 128);
    reg_head<<<dim3(1), dim3(NGRAPH), 0, stream>>>(dbuf1, (const float*)d_in[29],
                                                   (const float*)d_in[30], (float*)d_out, 128);
}
